// E29aSelectiveElmanCell_2147483648060
// MI455X (gfx1250) — compile-verified, matrix operand order checked by reading, not run on hardware
//
#include <hip/hip_runtime.h>
#include <math.h>
#include <stdint.h>

// ---------------------------------------------------------------------------
// E29a Selective Elman cell for MI455X (gfx1250).
// Phase 1: bf16 WMMA GEMM xz = x @ W_xz^T, 2x4 register-blocked tiles.
// Phase 2: persistent scan; 32 WGs each own a 32-col slice of D=1024.
//          W_h/W_write slices + tape slice resident in LDS (320KB/WGP).
//          Per-step h_work broadcast staged to LDS with async-to-LDS loads.
//          Recurrent (16x1024)x(1024x32) GEMMs via v_wmma_f32_16x16x32_bf16.
// ---------------------------------------------------------------------------

typedef __attribute__((ext_vector_type(16))) __bf16 v16bf;
typedef __attribute__((ext_vector_type(8)))  __bf16 v8bf;
typedef __attribute__((ext_vector_type(8)))  float  v8f;
typedef int v4i __attribute__((vector_size(4 * sizeof(int))));

#define BATCH 16
#define TSTEPS 512
#define DDIM 1024
#define NSLOT 16
#define NWG 32          // workgroups in scan kernel
#define SLICE 32        // D columns per workgroup
#define BTD (BATCH * TSTEPS * DDIM)

struct GridBar { unsigned cnt; unsigned gen; };

// ---------------- async global->LDS copy (CDNA5), guarded -------------------
#if defined(__has_builtin)
#if __has_builtin(__builtin_amdgcn_global_load_async_to_lds_b128)
#define HAVE_ASYNC_LDS 1
#endif
#endif
#ifndef HAVE_ASYNC_LDS
#define HAVE_ASYNC_LDS 0
#endif

__device__ __forceinline__ void async_copy_b128(void* ldst, const void* gsrc) {
#if HAVE_ASYNC_LDS
  // builtin signature on this toolchain: (v4i* global_src, v4i* lds_dst,
  // imm offset, imm cpol) with generic-typed pointers; keep provenance so the
  // backend can infer AS1/AS3.
  __builtin_amdgcn_global_load_async_to_lds_b128(
      (v4i*)const_cast<void*>(gsrc), (v4i*)ldst, 0, 0);
#else
  *(uint4*)ldst = *(const uint4*)gsrc;
#endif
}

__device__ __forceinline__ void async_copy_wait() {
#if HAVE_ASYNC_LDS
#if __has_builtin(__builtin_amdgcn_s_wait_asynccnt)
  __builtin_amdgcn_s_wait_asynccnt(0);
#else
  asm volatile("s_wait_asynccnt 0" ::: "memory");
#endif
#endif
}

// --- fragment load: 16-bit A/B layout (ISA 7.12.2). lane<16 -> K {0..7},
// lane>=16 -> K {8..15}; second half of VGPRs holds K+16. Row-major memory ->
// two contiguous b128 loads per lane (global or LDS).
__device__ __forceinline__ v16bf load_frag(const __bf16* p) {
  v8bf lo = *(const v8bf*)(p);
  v8bf hi = *(const v8bf*)(p + 16);
  v16bf r;
#pragma unroll
  for (int i = 0; i < 8; ++i) { r[i] = lo[i]; r[i + 8] = hi[i]; }
  return r;
}

// ---------------------------------------------------------------------------
__global__ void cvt_bf16_kernel(const float* __restrict__ src,
                                __bf16* __restrict__ dst, int n) {
  int i = blockIdx.x * blockDim.x + threadIdx.x;
  int stride = gridDim.x * blockDim.x;
  for (; i < n; i += stride) dst[i] = (__bf16)src[i];
}

__global__ void init_kernel(GridBar* bar, const float* __restrict__ hwinit,
                            __bf16* __restrict__ hwbuf) {
  int i = blockIdx.x * blockDim.x + threadIdx.x;
  if (i == 0) { bar->cnt = 0u; bar->gen = 0u; }
  int stride = gridDim.x * blockDim.x;
  for (int k = i; k < BATCH * DDIM; k += stride) hwbuf[k] = (__bf16)hwinit[k];
}

// ---------------------------------------------------------------------------
// xz GEMM: (B*T, D) x (D, 2D) -> stored time-major as XZ[t][b][e].
// Each wave: 2 row-tiles x 4 col-tiles (32x64 output), 8 WMMAs per K-step.
// ---------------------------------------------------------------------------
__global__ __launch_bounds__(256) void gemm_xz_kernel(
    const __bf16* __restrict__ Xb,   // (8192, 1024) row-major
    const __bf16* __restrict__ Wb,   // (2048, 1024) row-major (W_xz)
    float* __restrict__ XZ) {        // (512, 16, 2048)
  const int tid = threadIdx.x;
  const int wid = tid >> 5, lane = tid & 31;
  const int lrow = lane & 15, hi = lane >> 4;
  const int wtile = blockIdx.x * 8 + wid;      // 8192 wave-tiles
  const int mt2 = wtile >> 5;                  // 256 row super-tiles (32 rows)
  const int nt4 = wtile & 31;                  // 32 col super-tiles (64 cols)
  const int r0 = mt2 * 32, e0 = nt4 * 64;

  const __bf16* A0 = Xb + (size_t)(r0 + lrow) * DDIM;
  const __bf16* A1 = Xb + (size_t)(r0 + 16 + lrow) * DDIM;
  const __bf16* B0 = Wb + (size_t)(e0 + lrow) * DDIM;

  v8f acc[2][4] = {};
#pragma unroll 2
  for (int kk = 0; kk < 32; ++kk) {
    const int k0 = kk * 32 + hi * 8;
    v16bf a0 = load_frag(A0 + k0);
    v16bf a1 = load_frag(A1 + k0);
#pragma unroll
    for (int c = 0; c < 4; ++c) {
      v16bf b = load_frag(B0 + c * 16 * DDIM + k0);
      acc[0][c] = __builtin_amdgcn_wmma_f32_16x16x32_bf16(
          false, a0, false, b, (short)0, acc[0][c], false, false);
      acc[1][c] = __builtin_amdgcn_wmma_f32_16x16x32_bf16(
          false, a1, false, b, (short)0, acc[1][c], false, false);
    }
  }
  // C layout: lane<16 -> N=lane, M=i ; lane>=16 -> N=lane-16, M=i+8
  const int n = lrow, mbase = hi * 8;
#pragma unroll
  for (int m2 = 0; m2 < 2; ++m2)
#pragma unroll
    for (int c = 0; c < 4; ++c)
#pragma unroll
      for (int i = 0; i < 8; ++i) {
        const int r = r0 + m2 * 16 + mbase + i;   // r = b*512 + t
        const int bb = r >> 9, tt = r & 511;
        XZ[((size_t)tt * BATCH + bb) * (2 * DDIM) + e0 + c * 16 + n] =
            acc[m2][c][i];
      }
}

// ---------------------------------------------------------------------------
// grid-wide sense-reversing barrier (agent scope)
// ---------------------------------------------------------------------------
__device__ __forceinline__ void grid_barrier(GridBar* gb) {
  __syncthreads();
  if (threadIdx.x == 0) {
    __threadfence();
    unsigned g = __hip_atomic_load(&gb->gen, __ATOMIC_ACQUIRE,
                                   __HIP_MEMORY_SCOPE_AGENT);
    unsigned arrived = __hip_atomic_fetch_add(&gb->cnt, 1u, __ATOMIC_ACQ_REL,
                                              __HIP_MEMORY_SCOPE_AGENT);
    if (arrived == (unsigned)NWG - 1u) {
      __hip_atomic_store(&gb->cnt, 0u, __ATOMIC_RELAXED,
                         __HIP_MEMORY_SCOPE_AGENT);
      __hip_atomic_fetch_add(&gb->gen, 1u, __ATOMIC_RELEASE,
                             __HIP_MEMORY_SCOPE_AGENT);
    } else {
      while (__hip_atomic_load(&gb->gen, __ATOMIC_ACQUIRE,
                               __HIP_MEMORY_SCOPE_AGENT) == g) {
        __builtin_amdgcn_s_sleep(2);
      }
    }
    __threadfence();
  }
  __syncthreads();
}

__device__ __forceinline__ void softmax16(const float* s, float* a) {
  float m = s[0];
#pragma unroll
  for (int i = 1; i < 16; ++i) m = fmaxf(m, s[i]);
  float sum = 0.f;
#pragma unroll
  for (int i = 0; i < 16; ++i) { float e = __expf(s[i] - m); a[i] = e; sum += e; }
  const float inv = 1.f / sum;
#pragma unroll
  for (int i = 0; i < 16; ++i) a[i] *= inv;
}

// WG-level GEMM: (16 x 1024 bf16, LDS) x (1024 x 32 bf16 slice, LDS,
// stored as 32 rows of W i.e. [n][k]) -> 16x32 fp32 in lds_out.
// 8 waves: 2 col-tiles x 4 K-splits, LDS reduction.
__device__ __forceinline__ void wg_gemm(const __bf16* Al, const __bf16* Wl,
                                        float* lds_red, float* lds_out) {
  const int tid = threadIdx.x;
  const int wid = tid >> 5, lane = tid & 31;
  const int lrow = lane & 15, hi = lane >> 4;
  const int nt = wid & 1, ks = wid >> 1;
  v8f acc = {};
  const int kbase = ks * 256;
#pragma unroll
  for (int kk = 0; kk < 8; ++kk) {
    const int k0 = kbase + kk * 32 + hi * 8;
    v16bf a = load_frag(Al + lrow * DDIM + k0);
    v16bf b = load_frag(Wl + (nt * 16 + lrow) * DDIM + k0);
    acc = __builtin_amdgcn_wmma_f32_16x16x32_bf16(false, a, false, b,
                                                  (short)0, acc, false, false);
  }
#pragma unroll
  for (int i = 0; i < 8; ++i) lds_red[wid * 256 + lane * 8 + i] = acc[i];
  __syncthreads();
  if (wid < 2) {
    const int ntr = wid;
#pragma unroll
    for (int i = 0; i < 8; ++i) {
      float s = 0.f;
#pragma unroll
      for (int kr = 0; kr < 4; ++kr)
        s += lds_red[(kr * 2 + ntr) * 256 + lane * 8 + i];
      const int m = i + hi * 8;
      lds_out[m * SLICE + ntr * 16 + lrow] = s;
    }
  }
  __syncthreads();
}

// ---------------------------------------------------------------------------
// Persistent scan kernel. 32 WGs x 256 threads. WG j owns D columns
// [j*32, j*32+32).
// ---------------------------------------------------------------------------
__global__ __launch_bounds__(256) void scan_kernel(
    const float* __restrict__ xz,        // (T, B, 2D) fp32
    const float* __restrict__ bh,        // (D)
    const float* __restrict__ tape_init, // (B, N, D)
    const float* __restrict__ hw_init,   // (B, D)
    const __bf16* __restrict__ Wh_bf,    // (D, D)
    const __bf16* __restrict__ Ww_bf,    // (D, D)
    __bf16* __restrict__ hwbuf,          // 2 x (B, D) bf16 double buffer
    float* __restrict__ scoreA,          // (NWG, 256)
    float* __restrict__ scoreB,          // (NWG, 256)
    GridBar* bar,
    float* __restrict__ out) {           // output_all (B,T,D) ++ tape (B,N,D)
  extern __shared__ char smem[];
  __bf16* lds_Wh  = (__bf16*)(smem);                    // 65536 B
  __bf16* lds_Ww  = (__bf16*)(smem + 65536);            // 65536 B
  float* lds_tape = (float*)(smem + 131072);            // 32768 B
  __bf16* lds_hwA = (__bf16*)(smem + 163840);           // 32768 B (16x1024)
  float* lds_red  = (float*)(smem + 196608);            // 8192 B
  float* lds_hA   = (float*)(smem + 204800);            // 2048 B
  float* lds_hw   = (float*)(smem + 206848);            // 2048 B
  float* lds_rv   = (float*)(smem + 208896);            // 2048 B
  float* lds_wv   = (float*)(smem + 210944);            // 2048 B
  float* lds_sc   = (float*)(smem + 212992);            // 1024 B
  float* lds_attn = (float*)(smem + 214016);            // 1024 B

  const int tid = threadIdx.x;
  const int j = blockIdx.x;
  const float scale = 0.03125f;   // 1/sqrt(1024)

  // stage weight slices (rows [j*32, j*32+32) of W_h / W_write) via async DMA
  const char* WhS = (const char*)(Wh_bf + (size_t)j * SLICE * DDIM);
  const char* WwS = (const char*)(Ww_bf + (size_t)j * SLICE * DDIM);
  for (int idx = tid; idx < 4096; idx += 256) {
    async_copy_b128((char*)lds_Wh + idx * 16, WhS + idx * 16);
    async_copy_b128((char*)lds_Ww + idx * 16, WwS + idx * 16);
  }
  // stage h_work(0) (bf16, written by init_kernel) into LDS
  for (int idx = tid; idx < 2048; idx += 256)
    async_copy_b128((char*)lds_hwA + idx * 16, (const char*)hwbuf + idx * 16);
  // tape slice and fp32 h_work slice (strided gathers, plain loads)
  for (int idx = tid; idx < BATCH * NSLOT * SLICE; idx += 256) {
    const int bn = idx >> 5, dl = idx & 31;
    lds_tape[idx] = tape_init[(size_t)bn * DDIM + j * SLICE + dl];
  }
  for (int idx = tid; idx < BATCH * SLICE; idx += 256) {
    const int b = idx >> 5, dl = idx & 31;
    lds_hw[idx] = hw_init[(size_t)b * DDIM + j * SLICE + dl];
  }
  async_copy_wait();
  __syncthreads();

  for (int t = 0; t < TSTEPS; ++t) {
    __bf16* hwNext = hwbuf + (size_t)((t + 1) & 1) * (BATCH * DDIM);

    // ---- P1: recurrent GEMM (W_h, A already staged in LDS) + read scores ---
    wg_gemm(lds_hwA, lds_Wh, lds_red, lds_hA);
    {
      const int b = tid >> 4;
      float p = 0.f;
#pragma unroll 8
      for (int dl = 0; dl < SLICE; ++dl)
        p += lds_hw[b * SLICE + dl] * lds_tape[tid * SLICE + dl];
      scoreA[(size_t)j * 256 + tid] = p;
    }
    if (tid < 16 && t + 1 < TSTEPS)   // warm next step's xz rows
      __builtin_prefetch(&xz[(size_t)((t + 1) * BATCH + tid) * (2 * DDIM) +
                             j * SLICE], 0, 0);
    grid_barrier(bar);

    // ---- P2: softmax(read), read_val, h_work_new, output ----
    {
      float s = 0.f;
#pragma unroll 8
      for (int jj = 0; jj < NWG; ++jj) s += scoreA[(size_t)jj * 256 + tid];
      lds_sc[tid] = s * scale;
    }
    __syncthreads();
    if (tid < 16) softmax16(lds_sc + tid * 16, lds_attn + tid * 16);
    __syncthreads();
    for (int idx = tid; idx < BATCH * SLICE; idx += 256) {
      const int b = idx >> 5, dl = idx & 31, dg = j * SLICE + dl;
      float rv = 0.f;
#pragma unroll
      for (int n = 0; n < 16; ++n)
        rv += lds_attn[b * 16 + n] * lds_tape[(b * 16 + n) * SLICE + dl];
      lds_rv[idx] = rv;
      const size_t rowoff = (size_t)(t * BATCH + b) * (2 * DDIM);
      const float hn = tanhf(xz[rowoff + dg] + lds_hA[idx] + rv + bh[dg]);
      lds_hw[idx] = hn;
      hwNext[b * DDIM + dg] = (__bf16)hn;
      const float gpre = xz[rowoff + DDIM + dg] + rv + hn;
      const float gate = gpre / (1.f + __expf(-gpre));
      __builtin_nontemporal_store(hn * gate,
                                  &out[((size_t)b * TSTEPS + t) * DDIM + dg]);
    }
    grid_barrier(bar);

    // ---- P3: stage h_work_new (full 16x1024) to LDS; write GEMM + scores ---
    for (int idx = tid; idx < 2048; idx += 256)
      async_copy_b128((char*)lds_hwA + idx * 16,
                      (const char*)hwNext + idx * 16);
    async_copy_wait();
    __syncthreads();
    wg_gemm(lds_hwA, lds_Ww, lds_red, lds_wv);   // lds_hwA persists into P1(t+1)
    {
      const int b = tid >> 4;
      float p = 0.f;
#pragma unroll 8
      for (int dl = 0; dl < SLICE; ++dl)
        p += lds_wv[b * SLICE + dl] * lds_tape[tid * SLICE + dl];
      scoreB[(size_t)j * 256 + tid] = p;
    }
    grid_barrier(bar);

    // ---- P4: softmax(write), tape update (WG-local slice) ----
    {
      float s = 0.f;
#pragma unroll 8
      for (int jj = 0; jj < NWG; ++jj) s += scoreB[(size_t)jj * 256 + tid];
      lds_sc[tid] = s * scale;
    }
    __syncthreads();
    if (tid < 16) softmax16(lds_sc + tid * 16, lds_attn + tid * 16);
    __syncthreads();
    {
      const int b = tid >> 4;
      const float wa = lds_attn[tid];
#pragma unroll 8
      for (int dl = 0; dl < SLICE; ++dl) {
        const float tp = lds_tape[tid * SLICE + dl];
        lds_tape[tid * SLICE + dl] = tp + wa * (lds_wv[b * SLICE + dl] - tp);
      }
    }
    __syncthreads();
  }

  // final tape slice -> d_out tail (B,N,D)
  for (int idx = tid; idx < BATCH * NSLOT * SLICE; idx += 256) {
    const int bn = idx >> 5, dl = idx & 31;
    out[(size_t)BTD + (size_t)bn * DDIM + j * SLICE + dl] = lds_tape[idx];
  }
}

// ---------------------------------------------------------------------------
extern "C" void kernel_launch(void* const* d_in, const int* in_sizes, int n_in,
                              void* d_out, int out_size, void* d_ws,
                              size_t ws_size, hipStream_t stream) {
  const float* x        = (const float*)d_in[0];  // (16,512,1024)
  const float* tape0    = (const float*)d_in[1];  // (16,16,1024)
  const float* hw0      = (const float*)d_in[2];  // (16,1024)
  const float* W_h      = (const float*)d_in[3];  // (1024,1024)
  const float* W_xz     = (const float*)d_in[4];  // (2048,1024)
  const float* b_h      = (const float*)d_in[5];  // (1024)
  const float* W_write  = (const float*)d_in[6];  // (1024,1024)
  float* out = (float*)d_out;

  char* p = (char*)d_ws;
  GridBar* bar   = (GridBar*)p;            p += 256;
  float* scoreA  = (float*)p;              p += (size_t)NWG * 256 * 4;
  float* scoreB  = (float*)p;              p += (size_t)NWG * 256 * 4;
  __bf16* hwbuf  = (__bf16*)p;             p += (size_t)2 * BATCH * DDIM * 2;
  __bf16* x_bf   = (__bf16*)p;             p += (size_t)BATCH * TSTEPS * DDIM * 2;
  __bf16* wxz_bf = (__bf16*)p;             p += (size_t)2 * DDIM * DDIM * 2;
  __bf16* wh_bf  = (__bf16*)p;             p += (size_t)DDIM * DDIM * 2;
  __bf16* ww_bf  = (__bf16*)p;             p += (size_t)DDIM * DDIM * 2;
  float* xz      = (float*)p;              p += (size_t)BATCH * TSTEPS * 2 * DDIM * 4;

  cvt_bf16_kernel<<<4096, 256, 0, stream>>>(x, x_bf, BATCH * TSTEPS * DDIM);
  cvt_bf16_kernel<<<2048, 256, 0, stream>>>(W_xz, wxz_bf, 2 * DDIM * DDIM);
  cvt_bf16_kernel<<<1024, 256, 0, stream>>>(W_h, wh_bf, DDIM * DDIM);
  cvt_bf16_kernel<<<1024, 256, 0, stream>>>(W_write, ww_bf, DDIM * DDIM);
  init_kernel<<<64, 256, 0, stream>>>(bar, hw0, hwbuf);

  gemm_xz_kernel<<<1024, 256, 0, stream>>>(x_bf, wxz_bf, xz);

  const size_t smem = 215040;  // < 320KB WGP LDS
  scan_kernel<<<NWG, 256, smem, stream>>>(xz, b_h, tape0, hw0, wh_bf, ww_bf,
                                          hwbuf, scoreA, scoreB, bar, out);
}
